// CRFTokenClassificationModel_18494129176991
// MI455X (gfx1250) — compile-verified, matrix operand order checked by reading, not run on hardware
//
#include <hip/hip_runtime.h>
#include <hip/hip_bf16.h>

// ---------------------------------------------------------------------------
// Problem constants (fixed by the reference: B=32, S=2048, H=1024, L=32)
// ---------------------------------------------------------------------------
#define Bsz    32
#define Sseq   2048
#define Hdim   1024
#define Ldim   32
#define Mrows  (Bsz * Sseq)   // 65536
#define NCHUNK 16
#define CHUNK  (Sseq / NCHUNK)   // 128
#define LOG0   (-1.0e30f)

typedef __attribute__((ext_vector_type(16))) __bf16          v16bf;
typedef __attribute__((ext_vector_type(8)))  float           v8f;
typedef __attribute__((ext_vector_type(8)))  unsigned short  ushort8;
typedef __attribute__((ext_vector_type(4)))  float           float4v;

// f32 -> bf16 bits, round-to-nearest-even
static __device__ __forceinline__ unsigned short f2bf(float f) {
    unsigned int u = __float_as_uint(f);
    u += 0x7FFFu + ((u >> 16) & 1u);
    return (unsigned short)(u >> 16);
}

static __device__ __forceinline__ float lanebcast(float v, int i) {
    return __int_as_float(__builtin_amdgcn_readlane(__float_as_int(v), i));
}

static __device__ __forceinline__ float wave_max32(float v) {
#pragma unroll
    for (int m = 16; m >= 1; m >>= 1) v = fmaxf(v, __shfl_xor(v, m, 32));
    return v;
}
static __device__ __forceinline__ float wave_sum32(float v) {
#pragma unroll
    for (int m = 16; m >= 1; m >>= 1) v += __shfl_xor(v, m, 32);
    return v;
}
static __device__ __forceinline__ int wave_isum32(int v) {
#pragma unroll
    for (int m = 16; m >= 1; m >>= 1) v += __shfl_xor(v, m, 32);
    return v;
}

// ---------------------------------------------------------------------------
// Kernel 1: logits = hidden @ W + b   (bf16 WMMA, HBM-bound streaming)
// One wave -> 16x32 output tile (two v_wmma_f32_16x16x32_bf16 accumulators).
// W staged into LDS as bf16, transposed [n][k], in two 512-K halves (32 KB).
// ---------------------------------------------------------------------------
#define KSPLIT 512

__global__ void __launch_bounds__(256)
crf_logits_gemm_kernel(const float* __restrict__ hidden,
                       const float* __restrict__ Wg,
                       const float* __restrict__ bias,
                       float* __restrict__ outLogits) {
    __shared__ unsigned short WT[Ldim * KSPLIT];   // 32 KB, WT[n*KSPLIT + k]

    const int tid  = threadIdx.x;
    const int wave = tid >> 5;
    const int lane = tid & 31;

    const int tileM = blockIdx.x * 128 + wave * 16;   // 8 waves * 16 rows
    const int rowA  = tileM + (lane & 15);
    const int kb    = (lane < 16) ? 0 : 8;            // A/B fragment K-phase
    const int n0    = lane & 15;                      // column within 16-wide tile

    const float* __restrict__ arow = hidden + (size_t)rowA * Hdim;

    v8f acc0 = {};
    v8f acc1 = {};

    for (int kk = 0; kk < Hdim; kk += KSPLIT) {
        __syncthreads();
        // cooperative fill: W[k][n] (f32) -> WT[n][k] (bf16)
        for (int idx = tid; idx < Ldim * KSPLIT; idx += 256) {
            int n = idx >> 9;          // /KSPLIT
            int k = idx & (KSPLIT - 1);
            WT[n * KSPLIT + k] = f2bf(Wg[(size_t)(kk + k) * Ldim + n]);
        }
        __syncthreads();

        for (int kc = 0; kc < KSPLIT; kc += 32) {
            const float* ap = arow + kk + kc + kb;
            __builtin_prefetch(ap + 128, 0, 3);       // global_prefetch_b8, near

            // ---- A fragment: 16 bf16 per lane (K = kb..kb+7, kb+16..kb+23)
            float4v a0 = *(const float4v*)(ap);
            float4v a1 = *(const float4v*)(ap + 4);
            float4v a2 = *(const float4v*)(ap + 16);
            float4v a3 = *(const float4v*)(ap + 20);
            union { unsigned short s[16]; v16bf v; } af;
#pragma unroll
            for (int e = 0; e < 4; ++e) {
                af.s[e]      = f2bf(a0[e]);
                af.s[e + 4]  = f2bf(a1[e]);
                af.s[e + 8]  = f2bf(a2[e]);
                af.s[e + 12] = f2bf(a3[e]);
            }

            // ---- B fragments from LDS (ds_load_b128 pairs)
            union { ushort8 u[2]; v16bf v; } bf0, bf1;
            bf0.u[0] = *(const ushort8*)&WT[n0 * KSPLIT + kc + kb];
            bf0.u[1] = *(const ushort8*)&WT[n0 * KSPLIT + kc + kb + 16];
            bf1.u[0] = *(const ushort8*)&WT[(n0 + 16) * KSPLIT + kc + kb];
            bf1.u[1] = *(const ushort8*)&WT[(n0 + 16) * KSPLIT + kc + kb + 16];

            acc0 = __builtin_amdgcn_wmma_f32_16x16x32_bf16(
                       false, af.v, false, bf0.v, (short)0, acc0, false, false);
            acc1 = __builtin_amdgcn_wmma_f32_16x16x32_bf16(
                       false, af.v, false, bf1.v, (short)0, acc1, false, false);
        }
    }

    // ---- epilogue: bias + store (C layout: vgpr r -> M = r + 8*(lane>=16))
    const float b0 = bias[n0];
    const float b1 = bias[n0 + 16];
    const int   half = lane >> 4;
#pragma unroll
    for (int r = 0; r < 8; ++r) {
        int m = r + half * 8;
        float* orow = outLogits + (size_t)(tileM + m) * Ldim;
        orow[n0]      = acc0[r] + b0;
        orow[n0 + 16] = acc1[r] + b1;
    }
}

// ---------------------------------------------------------------------------
// Kernel 2a: chunked forward scan. Grid (NCHUNK, Bsz), 32 waves per block.
// Wave r scans chunk c starting from basis vector e_r, producing row r of the
// chunk's 32x32 log-transfer matrix. Exp-domain rescaled matvec per step:
// one vectorized v_exp_f32 + 32x (v_readlane + v_fmac).
// ---------------------------------------------------------------------------
__global__ void __launch_bounds__(1024)
crf_chunk_scan_kernel(const float* __restrict__ logits,
                      const int*   __restrict__ mask,
                      const float* __restrict__ trans,
                      float* __restrict__ wsM) {
    const int c = blockIdx.x;              // chunk
    const int b = blockIdx.y;              // batch
    const int r = threadIdx.x >> 5;        // row (basis vector) = wave id
    const int j = threadIdx.x & 31;        // label index = lane

    const float* __restrict__ em  = logits + (size_t)b * Sseq * Ldim;
    const int*   __restrict__ msk = mask   + b * Sseq;

    // column j of exp(T), in registers (T entries are small: no overflow)
    float expTcol[Ldim];
#pragma unroll
    for (int i = 0; i < Ldim; ++i) expTcol[i] = __expf(trans[i * Ldim + j]);

    const int t0 = (c == 0) ? 1 : c * CHUNK;
    const int t1 = (c + 1) * CHUNK;

    // first step applied to basis e_r directly (avoids log(0) init)
    float alpha = (msk[t0] > 0) ? (trans[r * Ldim + j] + em[t0 * Ldim + j])
                                : ((j == r) ? 0.f : LOG0);

    for (int t = t0 + 1; t < t1; ++t) {
        float cmax = wave_max32(alpha);
        float beta = __expf(alpha - cmax);
        float acc  = 0.f;
#pragma unroll
        for (int i = 0; i < Ldim; ++i)
            acc = fmaf(lanebcast(beta, i), expTcol[i], acc);
        float nxt = cmax + __logf(acc) + em[t * Ldim + j];
        alpha = (msk[t] > 0) ? nxt : alpha;
    }

    // store row r of chunk matrix: wsM[((b*NCHUNK + c)*32 + r)*32 + j]
    wsM[(((size_t)b * NCHUNK + c) * Ldim + r) * Ldim + j] = alpha;
}

// ---------------------------------------------------------------------------
// Kernel 2b: per-batch combine (16 log-matvecs) + numerator. 1 wave / batch.
// Chunk-matrix entries are large (sum ~128 emissions) -> two-pass logsumexp.
// ---------------------------------------------------------------------------
__global__ void __launch_bounds__(32)
crf_combine_kernel(const float* __restrict__ logits,
                   const int*   __restrict__ labels,
                   const int*   __restrict__ mask,
                   const float* __restrict__ trans,
                   const float* __restrict__ startt,
                   const float* __restrict__ endt,
                   const float* __restrict__ wsM,
                   float* __restrict__ llh) {
    const int b = blockIdx.x;
    const int j = threadIdx.x;

    const float* __restrict__ em  = logits + (size_t)b * Sseq * Ldim;
    const int*   __restrict__ lab = labels + b * Sseq;
    const int*   __restrict__ msk = mask   + b * Sseq;

    // ---------------- numerator (lanes partition timesteps) ----------------
    float part = 0.f;
    for (int t = 1 + j; t < Sseq; t += 32) {
        int ct = lab[t];
        int pt = lab[t - 1];
        part += (em[t * Ldim + ct] + trans[pt * Ldim + ct]) * (float)msk[t];
    }
    part = wave_sum32(part);

    int mc = 0;
    for (int t = j; t < Sseq; t += 32) mc += msk[t];
    mc = wave_isum32(mc);
    const int first = lab[0];
    const int last  = lab[mc - 1];
    const float numerator = startt[first] + em[first] + part + endt[last];

    // ---------------- fold chunk matrices ----------------------------------
    float alpha = startt[j] + em[j];               // init at t = 0
    for (int c = 0; c < NCHUNK; ++c) {
        const float* Mc = wsM + ((size_t)b * NCHUNK + c) * Ldim * Ldim;
        float vv[Ldim];
        float m = LOG0;
#pragma unroll
        for (int i = 0; i < Ldim; ++i) {
            vv[i] = lanebcast(alpha, i) + Mc[i * Ldim + j];
            m = fmaxf(m, vv[i]);
        }
        float s = 0.f;
#pragma unroll
        for (int i = 0; i < Ldim; ++i) s += __expf(vv[i] - m);
        alpha = m + __logf(s);
    }

    // denominator = logsumexp(alpha + end)
    float v = alpha + endt[j];
    float cmax = wave_max32(v);
    float s = wave_sum32(__expf(v - cmax));
    if (j == 0) llh[b] = numerator - (cmax + __logf(s));
}

// ---------------------------------------------------------------------------
// Fallback: monolithic single-wave-per-batch scan (if ws too small for wsM)
// ---------------------------------------------------------------------------
__global__ void __launch_bounds__(32)
crf_forward_kernel(const float* __restrict__ logits,
                   const int*   __restrict__ labels,
                   const int*   __restrict__ mask,
                   const float* __restrict__ trans,
                   const float* __restrict__ startt,
                   const float* __restrict__ endt,
                   float* __restrict__ llh) {
    const int b = blockIdx.x;
    const int j = threadIdx.x;

    const float* __restrict__ em  = logits + (size_t)b * Sseq * Ldim;
    const int*   __restrict__ lab = labels + b * Sseq;
    const int*   __restrict__ msk = mask   + b * Sseq;

    float expTcol[Ldim];
#pragma unroll
    for (int i = 0; i < Ldim; ++i) expTcol[i] = __expf(trans[i * Ldim + j]);

    float part = 0.f;
    for (int t = 1 + j; t < Sseq; t += 32) {
        int ct = lab[t];
        int pt = lab[t - 1];
        part += (em[t * Ldim + ct] + trans[pt * Ldim + ct]) * (float)msk[t];
    }
    part = wave_sum32(part);
    int mc = 0;
    for (int t = j; t < Sseq; t += 32) mc += msk[t];
    mc = wave_isum32(mc);
    const float numerator = startt[lab[0]] + em[lab[0]] + part + endt[lab[mc - 1]];

    float alpha = startt[j] + em[j];
    for (int t = 1; t < Sseq; ++t) {
        float cmax = wave_max32(alpha);
        float beta = __expf(alpha - cmax);
        float acc  = 0.f;
#pragma unroll
        for (int i = 0; i < Ldim; ++i)
            acc = fmaf(lanebcast(beta, i), expTcol[i], acc);
        float nxt = cmax + __logf(acc) + em[t * Ldim + j];
        alpha = (msk[t] > 0) ? nxt : alpha;
    }

    float v = alpha + endt[j];
    float cmax = wave_max32(v);
    float s = wave_sum32(__expf(v - cmax));
    if (j == 0) llh[b] = numerator - (cmax + __logf(s));
}

// ---------------------------------------------------------------------------
// Kernel 3: loss = -mean(llh)
// ---------------------------------------------------------------------------
__global__ void __launch_bounds__(32)
crf_loss_kernel(const float* __restrict__ llh, float* __restrict__ out) {
    float v = llh[threadIdx.x];
    v = wave_sum32(v);
    if (threadIdx.x == 0) out[0] = -v / (float)Bsz;
}

// ---------------------------------------------------------------------------
extern "C" void kernel_launch(void* const* d_in, const int* in_sizes, int n_in,
                              void* d_out, int out_size, void* d_ws, size_t ws_size,
                              hipStream_t stream) {
    const float* hidden = (const float*)d_in[0];
    const float* Wg     = (const float*)d_in[1];
    const float* bias   = (const float*)d_in[2];
    const float* trans  = (const float*)d_in[3];
    const float* startt = (const float*)d_in[4];
    const float* endt   = (const float*)d_in[5];
    const int*   labels = (const int*)d_in[6];
    const int*   mask   = (const int*)d_in[7];

    float* out    = (float*)d_out;
    float* loss   = out;          // element 0
    float* logits = out + 1;      // B*S*L elements

    const size_t nM = (size_t)Bsz * NCHUNK * Ldim * Ldim;  // 524288 floats (2MB)
    float* wsM = (float*)d_ws;
    float* llh = wsM + nM;

    // 1) logits GEMM: 65536 rows / (8 waves * 16 rows) = 512 blocks
    crf_logits_gemm_kernel<<<Mrows / 128, 256, 0, stream>>>(hidden, Wg, bias, logits);

    if (ws_size >= (nM + Bsz) * sizeof(float)) {
        // 2a) parallel chunk scans: 16K waves, 128 serial steps each
        crf_chunk_scan_kernel<<<dim3(NCHUNK, Bsz), 1024, 0, stream>>>(
            logits, mask, trans, wsM);
        // 2b) per-batch fold + numerator
        crf_combine_kernel<<<Bsz, 32, 0, stream>>>(logits, labels, mask, trans,
                                                   startt, endt, wsM, llh);
    } else {
        llh = (float*)d_ws;
        crf_forward_kernel<<<Bsz, 32, 0, stream>>>(logits, labels, mask, trans,
                                                   startt, endt, llh);
    }

    // 3) loss reduction
    crf_loss_kernel<<<1, 32, 0, stream>>>(llh, loss);
}
